// EGNNLayer_72146860638421
// MI455X (gfx1250) — compile-verified, matrix operand order checked by reading, not run on hardware
//
#include <hip/hip_runtime.h>
#include <hip/hip_bf16.h>

// Problem constants (fixed by the reference).
#define NN 10000
#define EE 320000
#define CC 256

typedef __attribute__((ext_vector_type(16))) __bf16 v16bf;
typedef __attribute__((ext_vector_type(8)))  __bf16 v8bf;
typedef __attribute__((ext_vector_type(8)))  float  v8f;
typedef __attribute__((ext_vector_type(4)))  int    v4i;

#if __has_builtin(__builtin_amdgcn_global_load_async_to_lds_b128) && \
    __has_builtin(__builtin_amdgcn_s_wait_asynccnt)
#define USE_ASYNC_LDS 1
typedef __attribute__((address_space(1))) v4i GV4i;   // global (AS1)
typedef __attribute__((address_space(3))) v4i LV4i;   // LDS (AS3)
#endif

__device__ __forceinline__ float silu_f(float x) {
    return x / (1.0f + __expf(-x));
}

__device__ __forceinline__ v16bf cat8(v8bf lo, v8bf hi) {
    return __builtin_shufflevector(lo, hi, 0,1,2,3,4,5,6,7,8,9,10,11,12,13,14,15);
}

__device__ __forceinline__ v8f wmma_bf16(v16bf a, v16bf b, v8f c) {
    return __builtin_amdgcn_wmma_f32_16x16x32_bf16(false, a, false, b,
                                                   (short)0, c, false, false);
}

// ---------------------------------------------------------------------------
// Pack a (Krows x 256) row-major f32 weight matrix into bf16 WMMA B-fragment
// layout for v_wmma_f32_16x16x32_bf16. K padded to KT*32 with zeros.
//   K = kt*32 + e + 8*((e>=8) + (lane>=16)),  N = nt*16 + (lane&15)
// Packed index = ((nt*KT + kt)*32 + lane)*16 + e  (== linear thread id t).
// ---------------------------------------------------------------------------
__global__ void pack_w_kernel(const float* __restrict__ W,
                              __bf16* __restrict__ out,
                              int Krows, int KT) {
    int t    = blockIdx.x * 256 + threadIdx.x;
    int e    = t & 15;
    int lane = (t >> 4) & 31;
    int frag = t >> 9;
    int kt   = frag % KT;
    int nt   = frag / KT;
    if (nt >= 16) return;
    int k = kt * 32 + e + (((e >> 3) + (lane >> 4)) << 3);
    int n = nt * 16 + (lane & 15);
    float v = (k < Krows) ? W[k * 256 + n] : 0.0f;
    out[t] = (__bf16)v;
}

// ---------------------------------------------------------------------------
// Init: zero m_agg, seed x_new = x, convert h to bf16 (for LDS-DMA gathers).
// ---------------------------------------------------------------------------
__global__ void init_kernel(float* __restrict__ m_agg,
                            float* __restrict__ x_new,
                            const float* __restrict__ x,
                            const float* __restrict__ h,
                            __bf16* __restrict__ hbf) {
    int t = blockIdx.x * 256 + threadIdx.x;
    if (t < NN * CC) {
        m_agg[t] = 0.0f;
        hbf[t]   = (__bf16)h[t];
    }
    if (t < NN * 3) x_new[t] = x[t];
}

// ---------------------------------------------------------------------------
// Edge kernel: one workgroup (256 threads / 8 waves) = 32 edges (2 M-tiles).
// Each B fragment (1KB) feeds two WMMAs -> 2x L2 weight-traffic reuse.
// Gather of h rows uses GLOBAL_LOAD_ASYNC_TO_LDS_B128 when available
// (one async instruction per 512B bf16 row), else manual b128 copies.
// ---------------------------------------------------------------------------
__global__ void __launch_bounds__(256)
edge_kernel(const __bf16* __restrict__ hbf, const float* __restrict__ x,
            const int* __restrict__ ei,
            const __bf16* __restrict__ W1p, const float* __restrict__ b1,
            const __bf16* __restrict__ W2p, const float* __restrict__ b2,
            const float* __restrict__ cW, const float* __restrict__ cb,
            float* __restrict__ m_agg, float* __restrict__ x_new) {
    __shared__ __bf16 s_feat[32][544];   // [h_row | h_col | dist^2 | 0-pad]
    __shared__ __bf16 s_m[32][272];
    __shared__ float  s_diff[32][3];
    __shared__ float  s_cw[32];
    __shared__ int    s_row[32];
    __shared__ int    s_col[32];

    const int t    = threadIdx.x;
    const int lane = t & 31;
    const int wave = t >> 5;
    const int e0   = blockIdx.x * 32;

    // ---- Edge metadata: indices, diff, dist^2, K-pad ---------------------
    if (t < 32) {
        const int e = e0 + t;
        const int r = ei[e];
        const int c = ei[EE + e];
        s_row[t] = r;
        s_col[t] = c;
        float dx = x[r * 3 + 0] - x[c * 3 + 0];
        float dy = x[r * 3 + 1] - x[c * 3 + 1];
        float dz = x[r * 3 + 2] - x[c * 3 + 2];
        s_diff[t][0] = dx; s_diff[t][1] = dy; s_diff[t][2] = dz;
        s_cw[t] = 0.0f;
        s_feat[t][512] = (__bf16)(dx * dx + dy * dy + dz * dz);
        #pragma unroll
        for (int z = 513; z < 544; ++z) s_feat[t][z] = (__bf16)0.0f;
    }
    __syncthreads();

    // ---- Gather 64 bf16 rows (32 x h[row], 32 x h[col]) into LDS ---------
    {
        #pragma unroll
        for (int q = 0; q < 8; ++q) {
            const int g    = wave * 8 + q;       // 0..63
            const int i    = g & 31;             // edge-in-tile
            const int half = g >> 5;             // 0: h[row], 1: h[col]
            const int idx  = half ? s_col[i] : s_row[i];
            const __bf16* src = hbf + (size_t)idx * CC + lane * 8;
            __bf16*       dst = &s_feat[i][half * 256 + lane * 8];
#ifdef USE_ASYNC_LDS
            __builtin_amdgcn_global_load_async_to_lds_b128(
                (GV4i*)src, (LV4i*)dst, 0, 0);
#else
            *(v8bf*)dst = *(const v8bf*)src;
#endif
        }
#ifdef USE_ASYNC_LDS
        __builtin_amdgcn_s_wait_asynccnt(0);
#endif
    }
    __syncthreads();

    const int m0  = lane & 15;         // A-row within M-tile
    const int mhi = (lane >> 4) << 3;  // K sub-chunk / C-row offset

    // ---- GEMM1: edge_input(32x544) @ eW1 -> SiLU -> s_m ------------------
    for (int nt = wave; nt < 16; nt += 8) {
        v8f acc0 = {}, acc1 = {};
        #pragma unroll
        for (int kt = 0; kt < 17; ++kt) {
            if (kt + 1 < 17)
                __builtin_prefetch(W1p + ((size_t)(nt * 17 + kt + 1) * 32 + lane) * 16, 0, 3);
            const int k0 = kt * 32 + mhi;
            v16bf b  = *(const v16bf*)(W1p + ((size_t)(nt * 17 + kt) * 32 + lane) * 16);
            v16bf a0 = cat8(*(const v8bf*)&s_feat[m0][k0],
                            *(const v8bf*)&s_feat[m0][k0 + 16]);
            v16bf a1 = cat8(*(const v8bf*)&s_feat[16 + m0][k0],
                            *(const v8bf*)&s_feat[16 + m0][k0 + 16]);
            acc0 = wmma_bf16(a0, b, acc0);
            acc1 = wmma_bf16(a1, b, acc1);
        }
        const int   n    = nt * 16 + (lane & 15);
        const float bias = b1[n];
        #pragma unroll
        for (int r = 0; r < 8; ++r) {
            s_m[mhi + r][n]      = (__bf16)silu_f(acc0[r] + bias);
            s_m[16 + mhi + r][n] = (__bf16)silu_f(acc1[r] + bias);
        }
    }
    __syncthreads();

    // ---- GEMM2: m(32x256) @ eW2 -> SiLU -> atomics -----------------------
    const float cb0 = cb[0];
    for (int nt = wave; nt < 16; nt += 8) {
        v8f acc0 = {}, acc1 = {};
        #pragma unroll
        for (int kt = 0; kt < 8; ++kt) {
            if (kt + 1 < 8)
                __builtin_prefetch(W2p + ((size_t)(nt * 8 + kt + 1) * 32 + lane) * 16, 0, 3);
            const int k0 = kt * 32 + mhi;
            v16bf b  = *(const v16bf*)(W2p + ((size_t)(nt * 8 + kt) * 32 + lane) * 16);
            v16bf a0 = cat8(*(const v8bf*)&s_m[m0][k0],
                            *(const v8bf*)&s_m[m0][k0 + 16]);
            v16bf a1 = cat8(*(const v8bf*)&s_m[16 + m0][k0],
                            *(const v8bf*)&s_m[16 + m0][k0 + 16]);
            acc0 = wmma_bf16(a0, b, acc0);
            acc1 = wmma_bf16(a1, b, acc1);
        }
        const int   n    = nt * 16 + (lane & 15);
        const float bias = b2[n];
        const float cwn  = cW[n];
        #pragma unroll
        for (int r = 0; r < 8; ++r) {
            const float mij0 = silu_f(acc0[r] + bias);
            const float mij1 = silu_f(acc1[r] + bias);
            const int   M0   = mhi + r;
            const int   M1   = 16 + mhi + r;
            unsafeAtomicAdd(&m_agg[(size_t)s_row[M0] * CC + n], mij0);
            unsafeAtomicAdd(&m_agg[(size_t)s_row[M1] * CC + n], mij1);
            atomicAdd(&s_cw[M0], mij0 * cwn);   // LDS ds_add_f32 partial dot
            atomicAdd(&s_cw[M1], mij1 * cwn);
        }
    }
    __syncthreads();

    // ---- Coordinate update: tanh(dot + cb) * diff, atomic into x_new -----
    if (t < 32) {
        const float w = tanhf(s_cw[t] + cb0);
        const int   r = s_row[t];
        unsafeAtomicAdd(&x_new[r * 3 + 0], s_diff[t][0] * w);
        unsafeAtomicAdd(&x_new[r * 3 + 1], s_diff[t][1] * w);
        unsafeAtomicAdd(&x_new[r * 3 + 2], s_diff[t][2] * w);
    }
}

// ---------------------------------------------------------------------------
// Node kernel: one workgroup = 16 nodes. [h | m_agg](16x512) @ nW1 -> SiLU
// -> @ nW2 + nb2 -> h_new (f32 stores, no atomics).  (3% of total FLOPs.)
// ---------------------------------------------------------------------------
__global__ void __launch_bounds__(256)
node_kernel(const float* __restrict__ h, const float* __restrict__ m_agg,
            const __bf16* __restrict__ W1p, const float* __restrict__ b1,
            const __bf16* __restrict__ W2p, const float* __restrict__ b2,
            float* __restrict__ h_new) {
    __shared__ __bf16 s_feat[16][544];
    __shared__ __bf16 s_m[16][272];

    const int t    = threadIdx.x;
    const int lane = t & 31;
    const int wave = t >> 5;
    const int n0   = blockIdx.x * 16;

    {
        const int i = t >> 4;
        const int j = t & 15;
        const int node = n0 + i;
        const float* hr = h     + (size_t)node * CC;
        const float* mr = m_agg + (size_t)node * CC;
        #pragma unroll
        for (int s = 0; s < 16; ++s) {
            s_feat[i][j + 16 * s]       = (__bf16)hr[j + 16 * s];
            s_feat[i][256 + j + 16 * s] = (__bf16)mr[j + 16 * s];
        }
    }
    __syncthreads();

    const int m0  = lane & 15;
    const int mhi = (lane >> 4) << 3;

    // GEMM1: 16x512x256
    for (int nt = wave; nt < 16; nt += 8) {
        v8f acc = {};
        #pragma unroll
        for (int kt = 0; kt < 16; ++kt) {
            if (kt + 1 < 16)
                __builtin_prefetch(W1p + ((size_t)(nt * 16 + kt + 1) * 32 + lane) * 16, 0, 3);
            const int k0 = kt * 32 + mhi;
            v16bf a = cat8(*(const v8bf*)&s_feat[m0][k0],
                           *(const v8bf*)&s_feat[m0][k0 + 16]);
            v16bf b = *(const v16bf*)(W1p + ((size_t)(nt * 16 + kt) * 32 + lane) * 16);
            acc = wmma_bf16(a, b, acc);
        }
        const int   n    = nt * 16 + (lane & 15);
        const float bias = b1[n];
        #pragma unroll
        for (int r = 0; r < 8; ++r) {
            s_m[mhi + r][n] = (__bf16)silu_f(acc[r] + bias);
        }
    }
    __syncthreads();

    // GEMM2: 16x256x256 -> h_new
    for (int nt = wave; nt < 16; nt += 8) {
        v8f acc = {};
        #pragma unroll
        for (int kt = 0; kt < 8; ++kt) {
            const int k0 = kt * 32 + mhi;
            v16bf a = cat8(*(const v8bf*)&s_m[m0][k0],
                           *(const v8bf*)&s_m[m0][k0 + 16]);
            v16bf b = *(const v16bf*)(W2p + ((size_t)(nt * 8 + kt) * 32 + lane) * 16);
            acc = wmma_bf16(a, b, acc);
        }
        const int   n    = nt * 16 + (lane & 15);
        const float bias = b2[n];
        #pragma unroll
        for (int r = 0; r < 8; ++r) {
            const int node = n0 + mhi + r;
            h_new[(size_t)node * CC + n] = acc[r] + bias;
        }
    }
}

// ---------------------------------------------------------------------------
extern "C" void kernel_launch(void* const* d_in, const int* in_sizes, int n_in,
                              void* d_out, int out_size, void* d_ws, size_t ws_size,
                              hipStream_t stream) {
    const float* h   = (const float*)d_in[0];
    const float* x   = (const float*)d_in[1];
    const int*   ei  = (const int*)d_in[2];
    const float* eW1 = (const float*)d_in[3];
    const float* eb1 = (const float*)d_in[4];
    const float* eW2 = (const float*)d_in[5];
    const float* eb2 = (const float*)d_in[6];
    const float* nW1 = (const float*)d_in[7];
    const float* nb1 = (const float*)d_in[8];
    const float* nW2 = (const float*)d_in[9];
    const float* nb2 = (const float*)d_in[10];
    const float* cW  = (const float*)d_in[11];
    const float* cb  = (const float*)d_in[12];

    float* h_new = (float*)d_out;
    float* x_new = h_new + (size_t)NN * CC;

    // Workspace: m_agg (f32 N*256) | h bf16 | 4 packed bf16 weight blocks.
    float*  m_agg = (float*)d_ws;
    __bf16* hbf   = (__bf16*)(m_agg + (size_t)NN * CC);  // N*256
    __bf16* eW1p  = hbf  + (size_t)NN * CC;              // 544*256
    __bf16* eW2p  = eW1p + 544 * 256;                    // 256*256
    __bf16* nW1p  = eW2p + 256 * 256;                    // 512*256
    __bf16* nW2p  = nW1p + 512 * 256;                    // 256*256

    init_kernel<<<NN, 256, 0, stream>>>(m_agg, x_new, x, h, hbf);

    pack_w_kernel<<<17 * 32, 256, 0, stream>>>(eW1, eW1p, 513, 17);
    pack_w_kernel<<< 8 * 32, 256, 0, stream>>>(eW2, eW2p, 256, 8);
    pack_w_kernel<<<16 * 32, 256, 0, stream>>>(nW1, nW1p, 512, 16);
    pack_w_kernel<<< 8 * 32, 256, 0, stream>>>(nW2, nW2p, 256, 8);

    edge_kernel<<<EE / 32, 256, 0, stream>>>(hbf, x, ei, eW1p, eb1, eW2p, eb2,
                                             cW, cb, m_agg, x_new);
    node_kernel<<<NN / 16, 256, 0, stream>>>(h, m_agg, nW1p, nb1, nW2p, nb2,
                                             h_new);
}
